// ToyModel_37151467110516
// MI455X (gfx1250) — compile-verified, matrix operand order checked by reading, not run on hardware
//
#include <hip/hip_runtime.h>
#include <hip/hip_bf16.h>
#include <math.h>

typedef __attribute__((ext_vector_type(16))) __bf16 v16bf;
typedef __attribute__((ext_vector_type(8)))  __bf16 v8bf;
typedef __attribute__((ext_vector_type(8)))  float  v8f;

constexpr int BATCH = 4, SEQ = 2048, EMBED = 256, HIDDEN = 512;
constexpr int BS = BATCH * SEQ;

// Load a 16x32 bf16 WMMA fragment from a row-major matrix R (16 rows used,
// leading dim ld). Works for both A (rows = M) and B when B is stored as
// Bt[n][k] (rows = N). Layout per CDNA5 ISA:
//   lane l (l<16):  row = l,    elements K = k0+0..7  and k0+16..23
//   lane l (l>=16): row = l-16, elements K = k0+8..15 and k0+24..31
__device__ __forceinline__ v16bf load_frag(const __bf16* R, int ld, int k0) {
  int lane = threadIdx.x & 31;
  const __bf16* p = R + (size_t)(lane & 15) * ld + (k0 + ((lane >> 4) << 3));
  v8bf lo = *(const v8bf*)p;
  v8bf hi = *(const v8bf*)(p + 16);
  v16bf f;
#pragma unroll
  for (int i = 0; i < 8; ++i) { f[i] = lo[i]; f[i + 8] = hi[i]; }
  return f;
}

__device__ __forceinline__ v8f wmma_bf16(v16bf a, v16bf b, v8f c) {
  return __builtin_amdgcn_wmma_f32_16x16x32_bf16(
      false, a, false, b, (short)0, c, false, false);
}

// ---- prep kernels ------------------------------------------------------

// Wt[n*K + k] = (bf16) W[k*N + n]   (transpose-convert weights)
__global__ void convert_wt_kernel(const float* __restrict__ W,
                                  __bf16* __restrict__ Wt, int K, int N) {
  int idx = blockIdx.x * 256 + threadIdx.x;  // N*K threads total
  int n = idx / K, k = idx % K;
  Wt[idx] = (__bf16)W[(size_t)k * N + n];
}

// X[i][e] = (bf16) emb_table[token_ids[i]][e]
__global__ void embed_kernel(const int* __restrict__ tok,
                             const float* __restrict__ emb,
                             __bf16* __restrict__ X) {
  int idx = blockIdx.x * 256 + threadIdx.x;  // BS*EMBED threads
  int row = idx >> 8;                        // EMBED == 256
  int e = idx & 255;
  X[idx] = (__bf16)emb[(size_t)tok[row] * EMBED + e];
}

// VT[b][h][s] = V[b][s][h]
__global__ void transpose_v_kernel(const __bf16* __restrict__ V,
                                   __bf16* __restrict__ VT) {
  int idx = blockIdx.x * 256 + threadIdx.x;  // BATCH*HIDDEN*SEQ threads
  int b = idx / (HIDDEN * SEQ);
  int rem = idx % (HIDDEN * SEQ);
  int h = rem / SEQ, s = rem % SEQ;
  VT[idx] = V[((size_t)b * SEQ + s) * HIDDEN + h];
}

// ---- WMMA GEMM: C(bf16)[M][512] = act(A[M][K] @ Bt[512][K]^T + bias) ---
// block = 256 threads (8 waves). Wave w owns a 16x64 slice (4 accumulators,
// A-fragment reused across 4 WMMAs per k-step). One block = one 16x512 strip.
// All 5 fragment loads are issued before any WMMA so the matrix pipe runs
// behind partial s_wait_loadcnt instead of draining the memory pipe.
__global__ void gemm_bf16_kernel(const __bf16* __restrict__ A,
                                 const __bf16* __restrict__ Bt,
                                 const float* __restrict__ bias,
                                 __bf16* __restrict__ C,
                                 int N, int K, int act) {
  int wave = threadIdx.x >> 5, lane = threadIdx.x & 31;
  int m0 = blockIdx.x << 4;
  int n0 = wave << 6;                      // 64 columns per wave (N == 512)
  const __bf16* Arow = A + (size_t)m0 * K;
  const __bf16* B0 = Bt + (size_t)(n0 +  0) * K;
  const __bf16* B1 = Bt + (size_t)(n0 + 16) * K;
  const __bf16* B2 = Bt + (size_t)(n0 + 32) * K;
  const __bf16* B3 = Bt + (size_t)(n0 + 48) * K;

  v8f acc[4] = {};
  for (int k0 = 0; k0 < K; k0 += 32) {
    v16bf a  = load_frag(Arow, K, k0);
    v16bf b0 = load_frag(B0, K, k0);
    v16bf b1 = load_frag(B1, K, k0);
    v16bf b2 = load_frag(B2, K, k0);
    v16bf b3 = load_frag(B3, K, k0);
    // prefetch next A chunk for this lane (global_prefetch_b8)
    const __bf16* pa =
        Arow + (size_t)(lane & 15) * K + k0 + ((lane >> 4) << 3);
    __builtin_prefetch(pa + 32, 0, 3);
    acc[0] = wmma_bf16(a, b0, acc[0]);
    acc[1] = wmma_bf16(a, b1, acc[1]);
    acc[2] = wmma_bf16(a, b2, acc[2]);
    acc[3] = wmma_bf16(a, b3, acc[3]);
  }

  int rbase = m0 + ((lane >> 4) << 3);
#pragma unroll
  for (int t = 0; t < 4; ++t) {
    int col = n0 + (t << 4) + (lane & 15);
    float bv = bias[col];
#pragma unroll
    for (int r = 0; r < 8; ++r) {
      float x = acc[t][r] + bv;
      if (act) x = 0.5f * x * (1.0f + erff(x * 0.70710678118654752f));
      C[(size_t)(rbase + r) * N + col] = (__bf16)x;
    }
  }
}

// ---- flash attention: 16 queries/block, 8 waves, 128-key steps ---------
// Phase 1: wave w computes the 16x16 score tile for keys j0+16w (K=512).
// Phase 2: 16 threads run the online-softmax update over 128 columns.
// Phase 3: every wave rescales and accumulates its 16x64 output slice,
//          P (16x128, LDS) @ V-slice (128x64) as 4x4 WMMAs.
__global__ void attention_kernel(const __bf16* __restrict__ Q,
                                 const __bf16* __restrict__ Km,
                                 const __bf16* __restrict__ VT,
                                 float* __restrict__ Out) {
  constexpr float scale = 0.04419417382415922f;  // 1/sqrt(512)
  int tid = threadIdx.x, wave = tid >> 5, lane = tid & 31;
  int q0 = blockIdx.x << 4;
  int b = blockIdx.y;

  __shared__ float sS[16][128];
  __shared__ __align__(16) __bf16 sP[16][128];
  __shared__ float sM[16], sL[16], sR[16];

  const __bf16* qb  = Q  + ((size_t)b * SEQ + q0) * HIDDEN;
  const __bf16* kb  = Km + (size_t)b * SEQ * HIDDEN;
  const __bf16* vtb = VT + (size_t)b * HIDDEN * SEQ;

  // hoist the 16x512 q-tile fragments (16 x v16bf = 128 VGPRs)
  v16bf qf[16];
#pragma unroll
  for (int h = 0; h < 16; ++h) qf[h] = load_frag(qb, HIDDEN, h << 5);

  if (tid < 16) { sM[tid] = -INFINITY; sL[tid] = 0.f; }
  v8f acc[4] = {};
  __syncthreads();

  for (int j0 = 0; j0 < SEQ; j0 += 128) {
    // phase 1: all 8 waves compute one 16x16 score tile each
    {
      v8f s = {};
      const __bf16* krow = kb + (size_t)(j0 + (wave << 4)) * HIDDEN;
#pragma unroll
      for (int h = 0; h < 16; ++h) {
        v16bf bb = load_frag(krow, HIDDEN, h << 5);
        s = wmma_bf16(qf[h], bb, s);
      }
      int col = (wave << 4) + (lane & 15);
      int rb = (lane >> 4) << 3;
#pragma unroll
      for (int r = 0; r < 8; ++r) sS[rb + r][col] = s[r] * scale;
    }
    __syncthreads();
    // phase 2: online softmax update, one thread per query row
    if (tid < 16) {
      float m_old = sM[tid];
      float m_new = m_old;
#pragma unroll
      for (int j = 0; j < 128; ++j) m_new = fmaxf(m_new, sS[tid][j]);
      float resc = expf(m_old - m_new);
      float lsum = sL[tid] * resc;
#pragma unroll
      for (int j = 0; j < 128; ++j) {
        float p = expf(sS[tid][j] - m_new);
        sP[tid][j] = (__bf16)p;
        lsum += p;
      }
      sM[tid] = m_new; sL[tid] = lsum; sR[tid] = resc;
    }
    __syncthreads();
    // phase 3: rescale accumulators, then O += P @ V-slice
    int rb = (lane >> 4) << 3;
    float rsc[8];
#pragma unroll
    for (int r = 0; r < 8; ++r) rsc[r] = sR[rb + r];
#pragma unroll
    for (int t = 0; t < 4; ++t)
#pragma unroll
      for (int r = 0; r < 8; ++r) acc[t][r] *= rsc[r];
#pragma unroll
    for (int k0 = 0; k0 < 128; k0 += 32) {
      // issue all loads for this k-step before the 4 WMMAs
      v16bf pf = load_frag(&sP[0][0], 128, k0);
      const __bf16* vb = vtb + (size_t)(wave << 6) * SEQ + j0 + k0;
      v16bf f0 = load_frag(vb + (size_t)0  * SEQ, SEQ, 0);
      v16bf f1 = load_frag(vb + (size_t)16 * SEQ, SEQ, 0);
      v16bf f2 = load_frag(vb + (size_t)32 * SEQ, SEQ, 0);
      v16bf f3 = load_frag(vb + (size_t)48 * SEQ, SEQ, 0);
      acc[0] = wmma_bf16(pf, f0, acc[0]);
      acc[1] = wmma_bf16(pf, f1, acc[1]);
      acc[2] = wmma_bf16(pf, f2, acc[2]);
      acc[3] = wmma_bf16(pf, f3, acc[3]);
    }
    __syncthreads();
  }

  float* outb = Out + ((size_t)b * SEQ + q0) * HIDDEN;
  int rb = (lane >> 4) << 3;
#pragma unroll
  for (int t = 0; t < 4; ++t) {
    int col = (wave << 6) + (t << 4) + (lane & 15);
#pragma unroll
    for (int r = 0; r < 8; ++r)
      outb[(size_t)(rb + r) * HIDDEN + col] = acc[t][r] / sL[rb + r];
  }
}

// ---- launch ------------------------------------------------------------

extern "C" void kernel_launch(void* const* d_in, const int* in_sizes, int n_in,
                              void* d_out, int out_size, void* d_ws, size_t ws_size,
                              hipStream_t stream) {
  (void)in_sizes; (void)n_in; (void)out_size; (void)ws_size;
  const int*   tok = (const int*)d_in[0];
  const float* emb = (const float*)d_in[1];
  const float* W1  = (const float*)d_in[2];
  const float* b1  = (const float*)d_in[3];
  const float* Wq  = (const float*)d_in[4];
  const float* bq  = (const float*)d_in[5];
  const float* Wk  = (const float*)d_in[6];
  const float* bk  = (const float*)d_in[7];
  const float* Wv  = (const float*)d_in[8];
  const float* bv  = (const float*)d_in[9];
  float* out = (float*)d_out;

  char* ws = (char*)d_ws;
  size_t o = 0;
  __bf16* X   = (__bf16*)(ws + o); o += (size_t)BS * EMBED * 2;        // 4 MB
  __bf16* W1t = (__bf16*)(ws + o); o += (size_t)HIDDEN * EMBED * 2;    // 256 KB
  __bf16* Wqt = (__bf16*)(ws + o); o += (size_t)HIDDEN * HIDDEN * 2;   // 512 KB
  __bf16* Wkt = (__bf16*)(ws + o); o += (size_t)HIDDEN * HIDDEN * 2;
  __bf16* Wvt = (__bf16*)(ws + o); o += (size_t)HIDDEN * HIDDEN * 2;
  __bf16* ACT = (__bf16*)(ws + o); o += (size_t)BS * HIDDEN * 2;       // 8 MB
  __bf16* Qb  = (__bf16*)(ws + o); o += (size_t)BS * HIDDEN * 2;
  __bf16* Kb  = (__bf16*)(ws + o); o += (size_t)BS * HIDDEN * 2;
  __bf16* Vb  = (__bf16*)(ws + o); o += (size_t)BS * HIDDEN * 2;
  __bf16* VTb = (__bf16*)(ws + o); o += (size_t)BS * HIDDEN * 2;       // total ~46 MB

  convert_wt_kernel<<<(HIDDEN * EMBED) / 256, 256, 0, stream>>>(W1, W1t, EMBED, HIDDEN);
  convert_wt_kernel<<<(HIDDEN * HIDDEN) / 256, 256, 0, stream>>>(Wq, Wqt, HIDDEN, HIDDEN);
  convert_wt_kernel<<<(HIDDEN * HIDDEN) / 256, 256, 0, stream>>>(Wk, Wkt, HIDDEN, HIDDEN);
  convert_wt_kernel<<<(HIDDEN * HIDDEN) / 256, 256, 0, stream>>>(Wv, Wvt, HIDDEN, HIDDEN);

  embed_kernel<<<(BS * EMBED) / 256, 256, 0, stream>>>(tok, emb, X);

  gemm_bf16_kernel<<<BS / 16, 256, 0, stream>>>(X, W1t, b1, ACT, HIDDEN, EMBED, 1);
  gemm_bf16_kernel<<<BS / 16, 256, 0, stream>>>(ACT, Wqt, bq, Qb, HIDDEN, HIDDEN, 0);
  gemm_bf16_kernel<<<BS / 16, 256, 0, stream>>>(ACT, Wkt, bk, Kb, HIDDEN, HIDDEN, 0);
  gemm_bf16_kernel<<<BS / 16, 256, 0, stream>>>(ACT, Wvt, bv, Vb, HIDDEN, HIDDEN, 0);

  transpose_v_kernel<<<(BATCH * HIDDEN * SEQ) / 256, 256, 0, stream>>>(Vb, VTb);

  dim3 ga(SEQ / 16, BATCH);
  attention_kernel<<<ga, 256, 0, stream>>>(Qb, Kb, VTb, out);
}
